// HoloLinear_79809082294771
// MI455X (gfx1250) — compile-verified, matrix operand order checked by reading, not run on hardware
//
#include <hip/hip_runtime.h>

#define IN_F   2048
#define OUT_F  2048
#define NNZ    1048576
#define BATCH  8192

#define BM 128
#define BN 128
#define BK 32
#define LDA (BK + 8)                 // padded LDS row in halves (80B): no bank conflicts, 16B aligned
#define TILE_HALVES (BM * LDA)       // 5120 halves
#define TILE_BYTES  (TILE_HALVES * 2) // 10240 B per tile buffer

typedef __attribute__((ext_vector_type(16))) _Float16     v16h;
typedef __attribute__((ext_vector_type(8)))  float        v8f;
typedef __attribute__((ext_vector_type(4)))  unsigned int u32x4;
typedef __attribute__((ext_vector_type(8)))  int          i32x8;
typedef __attribute__((ext_vector_type(4)))  int          i32x4;

// ---------------- Phase 1: build dense W ----------------

__global__ void holo_zero_f32(float* __restrict__ p, int n) {
    int i = blockIdx.x * blockDim.x + threadIdx.x;
    if (i < n) p[i] = 0.0f;
}

__global__ void holo_scatter(const _Float16* __restrict__ w,
                             const long long* __restrict__ coords,
                             float* __restrict__ W) {
    int i = blockIdx.x * blockDim.x + threadIdx.x;
    if (i < NNZ) {
        long long c = coords[i];
        int row = (int)(c >> 11);      // / IN_F  (2048 = 2^11)
        int col = (int)(c & 2047);     // % IN_F
        atomicAdd(&W[row * IN_F + col], (float)w[i]);
    }
}

__global__ void holo_f32_to_f16(const float* __restrict__ src,
                                _Float16* __restrict__ dst, int n) {
    int i = blockIdx.x * blockDim.x + threadIdx.x;
    if (i < n) dst[i] = (_Float16)src[i];
}

// ---------------- TDM tile load: 128 rows x 32 halves, row stride IN_F, LDS-padded to LDA ----------------

__device__ __forceinline__ void tdm_tile_load(unsigned lds_byte_addr,
                                              const _Float16* tile_ptr,
                                              unsigned tensor_rows) {
    unsigned long long ga = (unsigned long long)(uintptr_t)tile_ptr;

    u32x4 g0;
    g0[0] = 1u;                                   // [1:0] count=1 (valid user descriptor)
    g0[1] = lds_byte_addr;                        // [63:32] lds_addr
    g0[2] = (unsigned)ga;                         // [95:64] global_addr[31:0]
    g0[3] = (unsigned)(ga >> 32) | (2u << 30);    // [120:96] global_addr[56:32]; [127:126] type=2

    i32x8 g1;
    g1[0] = (1 << 16)                 // data_size: 2 bytes
          | (1 << 20)                 // pad_enable
          | (3 << 22)                 // pad_interval: 16 DWORDs (= 32 halves = one tile row)
          | (3 << 25);                // pad_amount: 4 DWORDs (= 8 halves) -> LDS row = 40 halves
    g1[1] = (int)((unsigned)IN_F << 16);          // [79:48]  tensor_dim0 = 2048
    g1[2] = (int)(tensor_rows << 16);             // [111:80] tensor_dim1
    g1[3] = (int)((unsigned)BK << 16);            // [127:112] tile_dim0 = 32
    g1[4] = BM;                                   // [143:128] tile_dim1 = 128; tile_dim2 = 0
    g1[5] = IN_F;                                 // [207:160] tensor_dim0_stride = 2048
    g1[6] = 0;
    g1[7] = 0;

    i32x4 z4 = {0, 0, 0, 0};                      // groups 2/3 unused (<= 2-D tensor)
    i32x8 z8 = {0, 0, 0, 0, 0, 0, 0, 0};
    __builtin_amdgcn_tensor_load_to_lds(g0, g1, z4, z4, z8, 0);
}

// ---------------- Phase 2: out = X16 (8192x2048) * W16^T (2048x2048) ----------------
// 256 threads = 8 waves; block tile 128x128; K-step 32; TDM double-buffered LDS.

__global__ __launch_bounds__(256)
void holo_gemm_wmma(const _Float16* __restrict__ X,
                    const _Float16* __restrict__ Wt,
                    float* __restrict__ out) {
    // 4 buffers: A0, A1, B0, B1 (10240 B each = 40 KB)
    __shared__ __align__(16) _Float16 lds[4 * TILE_HALVES];

    const int t    = threadIdx.x;
    const int lane = t & 31;
    const int wid  = t >> 5;
    const int wm   = wid & 3;     // 0..3 : 32-row M sub-tile
    const int wn   = wid >> 2;    // 0..1 : 64-col N sub-tile
    const int l16  = lane & 15;
    const int half = lane >> 4;

    const int mBase0 = blockIdx.y * BM;
    const int nBase0 = blockIdx.x * BN;

    const _Float16* aBase = X  + (size_t)mBase0 * IN_F;
    const _Float16* bBase = Wt + (size_t)nBase0 * IN_F;
    const unsigned  ldsBase = (unsigned)(unsigned long long)(uintptr_t)lds;

    v8f acc[2][4];
#pragma unroll
    for (int i = 0; i < 2; ++i)
#pragma unroll
        for (int j = 0; j < 4; ++j)
            acc[i][j] = (v8f){0.f, 0.f, 0.f, 0.f, 0.f, 0.f, 0.f, 0.f};

    const int NT = IN_F / BK;     // 64 K-tiles
    int buf = 0;

    if (t < 32) {                 // wave 0 drives the TDM
        tdm_tile_load(ldsBase + 0 * TILE_BYTES, aBase, BATCH);
        tdm_tile_load(ldsBase + 2 * TILE_BYTES, bBase, OUT_F);
    }

    for (int kt = 0; kt < NT; ++kt) {
        if (t < 32) {
            if (kt + 1 < NT) {
                const int nb  = buf ^ 1;
                const int k0n = (kt + 1) * BK;
                tdm_tile_load(ldsBase + (0 + nb) * TILE_BYTES, aBase + k0n, BATCH);
                tdm_tile_load(ldsBase + (2 + nb) * TILE_BYTES, bBase + k0n, OUT_F);
                __builtin_amdgcn_s_wait_tensorcnt(2);  // in-order: current tile's 2 ops done
            } else {
                __builtin_amdgcn_s_wait_tensorcnt(0);
            }
        }
        __syncthreads();          // LDS[buf] visible to all waves

        const _Float16* Asb = lds + (0 + buf) * TILE_HALVES;
        const _Float16* Bsb = lds + (2 + buf) * TILE_HALVES;

        // A fragment (16x32 f16): lane<16 -> K 0..7 & 16..23, lane>=16 -> K 8..15 & 24..31.
        v16h a[2], b[4];
#pragma unroll
        for (int i = 0; i < 2; ++i) {
            const _Float16* p = &Asb[(wm * 32 + i * 16 + l16) * LDA];
            uint4* d = (uint4*)&a[i];
            d[0] = *(const uint4*)&p[half * 8];
            d[1] = *(const uint4*)&p[16 + half * 8];
        }
        // B(k,n) = W(n,k): per-lane pattern mirrors A with lane -> N.
#pragma unroll
        for (int j = 0; j < 4; ++j) {
            const _Float16* p = &Bsb[(wn * 64 + j * 16 + l16) * LDA];
            uint4* d = (uint4*)&b[j];
            d[0] = *(const uint4*)&p[half * 8];
            d[1] = *(const uint4*)&p[16 + half * 8];
        }

#pragma unroll
        for (int i = 0; i < 2; ++i)
#pragma unroll
            for (int j = 0; j < 4; ++j)
                acc[i][j] = __builtin_amdgcn_wmma_f32_16x16x32_f16(
                    false, a[i], false, b[j], (short)0, acc[i][j], false, false);

        __syncthreads();          // all waves done reading buf before TDM refills it
        buf ^= 1;
    }

    // Epilogue: C/D layout — VGPR v: lanes 0-15 -> M=v, lanes 16-31 -> M=v+8; N = lane%16.
#pragma unroll
    for (int i = 0; i < 2; ++i) {
        int mB = mBase0 + wm * 32 + i * 16 + half * 8;
#pragma unroll
        for (int j = 0; j < 4; ++j) {
            int n = nBase0 + wn * 64 + j * 16 + l16;
#pragma unroll
            for (int v = 0; v < 8; ++v)
                out[(size_t)(mB + v) * OUT_F + n] = acc[i][j][v];
        }
    }
}

// ---------------- Driver ----------------

extern "C" void kernel_launch(void* const* d_in, const int* in_sizes, int n_in,
                              void* d_out, int out_size, void* d_ws, size_t ws_size,
                              hipStream_t stream) {
    const float*     x      = (const float*)d_in[0];
    const _Float16*  w      = (const _Float16*)d_in[1];
    const long long* coords = (const long long*)d_in[2];
    float*           out    = (float*)d_out;

    char* ws = (char*)d_ws;
    float*    Wacc = (float*)ws;                                   // 16 MB fp32 accumulate
    _Float16* W16  = (_Float16*)(ws + (size_t)OUT_F * IN_F * 4);   //  8 MB f16 weights
    _Float16* X16  = (_Float16*)(ws + (size_t)OUT_F * IN_F * 6);   // 32 MB f16 activations

    const int nW = OUT_F * IN_F;
    const int nX = BATCH * IN_F;

    holo_zero_f32 <<<(nW  + 255) / 256, 256, 0, stream>>>(Wacc, nW);
    holo_scatter  <<<(NNZ + 255) / 256, 256, 0, stream>>>(w, coords, Wacc);
    holo_f32_to_f16<<<(nW + 255) / 256, 256, 0, stream>>>(Wacc, W16, nW);
    holo_f32_to_f16<<<(nX + 255) / 256, 256, 0, stream>>>(x, X16, nX);

    dim3 grid(OUT_F / BN, BATCH / BM);   // 16 x 64 = 1024 blocks
    holo_gemm_wmma<<<grid, 256, 0, stream>>>(X16, W16, out);
}
